// LinkPrediction_86706799772291
// MI455X (gfx1250) — matrix operand order repaired and checked by reading, not compiled, not run on hardware
//
#include <hip/hip_runtime.h>

// ---------------------------------------------------------------------------
// R-GCN (basis decomposition) x2 + DistMult link prediction for gfx1250.
// Strategy: counting-sort edges by relation so each 64-edge tile shares one
// W_r; stage W_r^T (bf16, 8KB) in LDS; per wave gather 16 src rows in the
// native 16-bit A-matrix lane layout and run v_wmma_f32_16x16x32_bf16
// (K=64 -> 2 wmma per N-tile, 4 N-tiles); scatter f32 messages with
// global_atomic_add_f32 into an L2-resident (51MB) aggregator.
// ---------------------------------------------------------------------------

typedef __attribute__((ext_vector_type(16))) __bf16 v16bf;
typedef __attribute__((ext_vector_type(8)))  float  v8f;

union ABPack {
    uint4 q[2];
    v16bf v;
};

static __device__ __forceinline__ unsigned short f32_to_bf16(float f) {
    union { float f; unsigned u; } x;
    x.f = f;
    unsigned u = x.u;
    if ((u & 0x7fffffffu) > 0x7f800000u)            // NaN: keep quiet NaN
        return (unsigned short)((u >> 16) | 0x0040u);
    unsigned r = u + 0x7fffu + ((u >> 16) & 1u);    // round-to-nearest-even
    return (unsigned short)(r >> 16);
}

// ---------------- sort edges by relation (counting sort, padded bins) ------

__global__ void hist_kernel(const int* __restrict__ ety, int E,
                            int* __restrict__ counts) {
    int i = blockIdx.x * blockDim.x + threadIdx.x;
    if (i < E) atomicAdd(&counts[ety[i]], 1);
}

__global__ void scan_kernel(const int* __restrict__ counts, int R, int tile,
                            int* __restrict__ off) {
    if (blockIdx.x == 0 && threadIdx.x == 0) {
        int acc = 0;
        for (int r = 0; r < R; ++r) {
            off[r] = acc;
            acc += ((counts[r] + tile - 1) / tile) * tile;   // pad bin to tile
        }
        off[R] = acc;
    }
}

__global__ void scatter_kernel(const int* __restrict__ src,
                               const int* __restrict__ dst,
                               const int* __restrict__ ety, int E,
                               const int* __restrict__ off,
                               int* __restrict__ cursor,
                               int* __restrict__ src_s,
                               int* __restrict__ dst_s) {
    int i = blockIdx.x * blockDim.x + threadIdx.x;
    if (i >= E) return;
    int r = ety[i];
    int pos = off[r] + atomicAdd(&cursor[r], 1);
    src_s[pos] = src[i];
    dst_s[pos] = dst[i];
}

// ---------------- f32 -> bf16 feature conversion ---------------------------

__global__ void cvt_kernel(const float* __restrict__ in,
                           unsigned short* __restrict__ out, int n) {
    int i = blockIdx.x * blockDim.x + threadIdx.x;
    if (i < n) out[i] = f32_to_bf16(in[i]);
}

// ---------------- fold bases: WrT[r][n][k] = sum_b c[r,b] * V[b][k][n] -----

__global__ void build_wrt(const float* __restrict__ V,
                          const float* __restrict__ C, int B,
                          unsigned short* __restrict__ WrT) {
    int r = blockIdx.x;
    for (int idx = threadIdx.x; idx < 64 * 64; idx += blockDim.x) {
        int n = idx >> 6, k = idx & 63;
        float acc = 0.f;
        #pragma unroll 4
        for (int b = 0; b < B; ++b)
            acc += C[r * B + b] * V[(size_t)b * 4096 + k * 64 + n];
        WrT[(size_t)r * 4096 + n * 64 + k] = f32_to_bf16(acc);
    }
}

__global__ void build_wlt(const float* __restrict__ Wl,
                          unsigned short* __restrict__ WlT) {
    int idx = blockIdx.x * blockDim.x + threadIdx.x;
    if (idx < 64 * 64) {
        int n = idx >> 6, k = idx & 63;
        WlT[n * 64 + k] = f32_to_bf16(Wl[k * 64 + n]);
    }
}

// ---------------- edge message kernel (WMMA + atomic scatter) --------------
// Block = 128 threads = 4 waves; each wave = 16 edges; block tile = 64 edges
// of a single relation (guaranteed by padded counting sort).

__global__ void edge_wmma(const unsigned short* __restrict__ hbf,
                          const unsigned short* __restrict__ WrT,
                          const int* __restrict__ src_s,
                          const int* __restrict__ dst_s,
                          const int* __restrict__ off, int R,
                          float* __restrict__ agg) {
    __shared__ unsigned short Wt[64 * 64];          // W_r^T, bf16, 8KB

    const int total = off[R];
    const int tileBase = blockIdx.x * 64;
    if (tileBase >= total) return;

    int r = 0;                                      // bin lookup (33 ints, L2)
    while (r + 1 < R && off[r + 1] <= tileBase) ++r;

    {   // stage W_r^T into LDS
        const uint4* g = (const uint4*)(WrT + (size_t)r * 4096);
        uint4* l = (uint4*)Wt;
        for (int i = threadIdx.x; i < 512; i += blockDim.x) l[i] = g[i];
    }
    __syncthreads();

    const int wave = threadIdx.x >> 5;
    const int lane = threadIdx.x & 31;
    const int m    = lane & 15;
    const int half = lane >> 4;                     // 0: K 0-7/16-23, 1: 8-15/24-31
    const int eb   = tileBase + wave * 16;

    // ---- gather A: 16 src rows, ISA 16-bit A-matrix lane layout ----------
    const int  e     = eb + m;
    const int  row   = src_s[e];
    const int  klo   = half * 8;
    ABPack a0, a1;
    if (row >= 0) {
        const uint4* rp = (const uint4*)(hbf + (size_t)row * 64);
        a0.q[0] = rp[(klo +  0) >> 3];
        a0.q[1] = rp[(klo + 16) >> 3];
        a1.q[0] = rp[(klo + 32) >> 3];
        a1.q[1] = rp[(klo + 48) >> 3];
    } else {                                        // padding slot -> zero row
        a0.q[0] = a0.q[1] = a1.q[0] = a1.q[1] = make_uint4(0u, 0u, 0u, 0u);
    }

    // ---- 8x v_wmma_f32_16x16x32_bf16: (16 x 64) @ (64 x 64) --------------
    v8f acc[4] = {};
    #pragma unroll
    for (int nj = 0; nj < 4; ++nj) {
        const int n = nj * 16 + m;                  // B lane column
        const unsigned short* wp = Wt + n * 64 + half * 8;
        ABPack b0, b1;
        b0.q[0] = ((const uint4*)wp)[0];            // K base .. +7
        b0.q[1] = ((const uint4*)(wp + 16))[0];     // K base+16 .. +23
        b1.q[0] = ((const uint4*)(wp + 32))[0];
        b1.q[1] = ((const uint4*)(wp + 48))[0];
        acc[nj] = __builtin_amdgcn_wmma_f32_16x16x32_bf16(
            false, a0.v, false, b0.v, (short)0, acc[nj], false, false);
        acc[nj] = __builtin_amdgcn_wmma_f32_16x16x32_bf16(
            false, a1.v, false, b1.v, (short)0, acc[nj], false, false);
    }

    // ---- scatter messages: C/D layout row = j + half*8, col = nj*16 + m --
    #pragma unroll
    for (int j = 0; j < 8; ++j) {
        const int em = eb + j + half * 8;
        const int d  = dst_s[em];
        if (d >= 0) {
            float* outp = agg + (size_t)d * 64 + m;
            #pragma unroll
            for (int nj = 0; nj < 4; ++nj)
                atomicAdd(outp + nj * 16, acc[nj][j]);
        }
    }
}

// ---------------- finalize: hout = act(agg + hin @ W_loop + bias) ----------

__global__ void finalize_k(const unsigned short* __restrict__ hin_bf,
                           const unsigned short* __restrict__ WlT,
                           const float* __restrict__ agg,
                           const float* __restrict__ bias, int relu,
                           float* __restrict__ hout,
                           unsigned short* hout_bf, int N) {
    __shared__ unsigned short Wt[64 * 64];
    {
        const uint4* g = (const uint4*)WlT;
        uint4* l = (uint4*)Wt;
        for (int i = threadIdx.x; i < 512; i += blockDim.x) l[i] = g[i];
    }
    __syncthreads();

    const int wave = threadIdx.x >> 5;
    const int lane = threadIdx.x & 31;
    const int m    = lane & 15;
    const int half = lane >> 4;
    const int base = blockIdx.x * 64 + wave * 16;
    if (base >= N) return;

    const int row = base + m;
    const int klo = half * 8;
    const uint4* rp = (const uint4*)(hin_bf + (size_t)row * 64);
    ABPack a0, a1;
    a0.q[0] = rp[(klo +  0) >> 3];
    a0.q[1] = rp[(klo + 16) >> 3];
    a1.q[0] = rp[(klo + 32) >> 3];
    a1.q[1] = rp[(klo + 48) >> 3];

    v8f acc[4] = {};
    #pragma unroll
    for (int nj = 0; nj < 4; ++nj) {
        const int n = nj * 16 + m;
        const unsigned short* wp = Wt + n * 64 + half * 8;
        ABPack b0, b1;
        b0.q[0] = ((const uint4*)wp)[0];
        b0.q[1] = ((const uint4*)(wp + 16))[0];
        b1.q[0] = ((const uint4*)(wp + 32))[0];
        b1.q[1] = ((const uint4*)(wp + 48))[0];
        acc[nj] = __builtin_amdgcn_wmma_f32_16x16x32_bf16(
            false, a0.v, false, b0.v, (short)0, acc[nj], false, false);
        acc[nj] = __builtin_amdgcn_wmma_f32_16x16x32_bf16(
            false, a1.v, false, b1.v, (short)0, acc[nj], false, false);
    }

    #pragma unroll
    for (int nj = 0; nj < 4; ++nj) {
        const int col = nj * 16 + m;
        const float bv = bias[col];
        #pragma unroll
        for (int j = 0; j < 8; ++j) {
            const int rr = base + j + half * 8;     // wave-exclusive rows
            float v = acc[nj][j] + agg[(size_t)rr * 64 + col] + bv;
            if (relu) v = v > 0.f ? v : 0.f;
            hout[(size_t)rr * 64 + col] = v;
            if (hout_bf) hout_bf[(size_t)rr * 64 + col] = f32_to_bf16(v);
        }
    }
}

// ---------------- DistMult triplet scoring ---------------------------------

__global__ void distmult_k(const float* __restrict__ h,
                           const float* __restrict__ wrel,
                           const int* __restrict__ head,
                           const int* __restrict__ rel,
                           const int* __restrict__ tail,
                           float* __restrict__ out, int T) {
    int t = blockIdx.x * blockDim.x + threadIdx.x;
    if (t >= T) return;
    const float4* ph = (const float4*)(h    + (size_t)head[t] * 64);
    const float4* pr = (const float4*)(wrel + (size_t)rel[t]  * 64);
    const float4* pt = (const float4*)(h    + (size_t)tail[t] * 64);
    float s = 0.f;
    #pragma unroll
    for (int i = 0; i < 16; ++i) {
        float4 a = ph[i], b = pr[i], c = pt[i];
        s += a.x * b.x * c.x + a.y * b.y * c.y
           + a.z * b.z * c.z + a.w * b.w * c.w;
    }
    out[t] = s;
}

// ---------------------------------------------------------------------------

extern "C" void kernel_launch(void* const* d_in, const int* in_sizes, int n_in,
                              void* d_out, int out_size, void* d_ws, size_t ws_size,
                              hipStream_t stream) {
    const int*   src   = (const int*)d_in[0];
    const int*   dst   = (const int*)d_in[1];
    const int*   ety   = (const int*)d_in[2];
    const int*   head  = (const int*)d_in[3];
    const int*   rel   = (const int*)d_in[4];
    const int*   tail  = (const int*)d_in[5];
    const float* embed = (const float*)d_in[6];
    const float* v0    = (const float*)d_in[7];
    const float* c0    = (const float*)d_in[8];
    const float* wl0   = (const float*)d_in[9];
    const float* b0    = (const float*)d_in[10];
    const float* v1    = (const float*)d_in[11];
    const float* c1    = (const float*)d_in[12];
    const float* wl1   = (const float*)d_in[13];
    const float* b1    = (const float*)d_in[14];
    const float* wrel  = (const float*)d_in[15];

    const int E = in_sizes[0];
    const int T = in_sizes[3];
    const int D = 64;
    const int N = in_sizes[6] / D;
    const int R = in_sizes[15] / D;
    const int B = in_sizes[8] / R;
    const int TILE = 64;
    const int Epad = E + R * TILE;
    const int maxTiles = (E + TILE - 1) / TILE + R;

    // workspace carve-out (256B aligned)
    char* wsb = (char*)d_ws;
    size_t o = 0;
    auto take = [&](size_t bytes) -> char* {
        o = (o + 255) & ~(size_t)255;
        char* p = wsb + o;
        o += bytes;
        return p;
    };
    int*            counts = (int*)take((size_t)R * 4);
    int*            off    = (int*)take((size_t)(R + 1) * 4);
    int*            cursor = (int*)take((size_t)R * 4);
    int*            src_s  = (int*)take((size_t)Epad * 4);
    int*            dst_s  = (int*)take((size_t)Epad * 4);
    unsigned short* hbf    = (unsigned short*)take((size_t)N * D * 2);
    float*          bufA   = (float*)take((size_t)N * D * 4);   // agg / final h
    float*          bufB   = (float*)take((size_t)N * D * 4);   // layer-0 out
    unsigned short* WrT    = (unsigned short*)take((size_t)R * D * D * 2);
    unsigned short* WlT    = (unsigned short*)take((size_t)D * D * 2);
    (void)ws_size; (void)n_in; (void)out_size;

    // ---- counting sort of edges by relation (done once, reused) ----------
    hipMemsetAsync(counts, 0,    (size_t)R * 4, stream);
    hipMemsetAsync(cursor, 0,    (size_t)R * 4, stream);
    hipMemsetAsync(src_s,  0xFF, (size_t)Epad * 4, stream);     // -1 padding
    hipMemsetAsync(dst_s,  0xFF, (size_t)Epad * 4, stream);

    hist_kernel   <<<(E + 255) / 256, 256, 0, stream>>>(ety, E, counts);
    scan_kernel   <<<1, 1, 0, stream>>>(counts, R, TILE, off);
    scatter_kernel<<<(E + 255) / 256, 256, 0, stream>>>(src, dst, ety, E, off,
                                                        cursor, src_s, dst_s);

    // ---- bf16 copy of input embeddings ------------------------------------
    cvt_kernel<<<(N * D + 255) / 256, 256, 0, stream>>>(embed, hbf, N * D);

    // ---- two R-GCN layers --------------------------------------------------
    for (int layer = 0; layer < 2; ++layer) {
        const float* V    = layer ? v1  : v0;
        const float* C    = layer ? c1  : c0;
        const float* Wl   = layer ? wl1 : wl0;
        const float* bias = layer ? b1  : b0;

        build_wrt<<<R, 256, 0, stream>>>(V, C, B, WrT);
        build_wlt<<<(D * D + 255) / 256, 256, 0, stream>>>(Wl, WlT);

        hipMemsetAsync(bufA, 0, (size_t)N * D * 4, stream);     // zero agg
        edge_wmma<<<maxTiles, 128, 0, stream>>>(hbf, WrT, src_s, dst_s, off,
                                                R, bufA);
        if (layer == 0) {
            // ReLU; refresh bf16 features in place (wave-exclusive rows)
            finalize_k<<<(N + 63) / 64, 128, 0, stream>>>(
                hbf, WlT, bufA, bias, 1, bufB, hbf, N);
        } else {
            // no activation; final f32 h lands in bufA (in-place over agg)
            finalize_k<<<(N + 63) / 64, 128, 0, stream>>>(
                hbf, WlT, bufA, bias, 0, bufA, (unsigned short*)nullptr, N);
        }
    }

    // ---- DistMult scores ---------------------------------------------------
    distmult_k<<<(T + 255) / 256, 256, 0, stream>>>(bufA, wrel, head, rel,
                                                    tail, (float*)d_out, T);
}
